// Attention_73048803771007
// MI455X (gfx1250) — matrix operand order repaired and checked by reading, not compiled, not run on hardware
//
#include <hip/hip_runtime.h>

typedef __bf16 bf16;
typedef __attribute__((ext_vector_type(16))) __bf16 v16bf;
typedef __attribute__((ext_vector_type(8)))  __bf16 v8bf;
typedef __attribute__((ext_vector_type(8)))  float  v8f;

#define DIM 768
#define NH  12
#define HD  64
#define BB  8
#define NN  1024

__device__ __forceinline__ v8f wmma_bf16(v16bf a, v16bf b, v8f c) {
  return __builtin_amdgcn_wmma_f32_16x16x32_bf16(false, a, false, b, (short)0, c, false, false);
}

// A-fragment (16x32 bf16, row-major source): lane holds row (lane&15);
// elements 0..15 -> K = kbase + hi*16 + i
__device__ __forceinline__ v16bf load_a_frag(const bf16* __restrict__ rowp, int kbase, int hi) {
  v8bf lo = *(const v8bf*)(rowp + kbase + hi * 16);
  v8bf hv = *(const v8bf*)(rowp + kbase + hi * 16 + 8);
  v16bf a;
#pragma unroll
  for (int i = 0; i < 8; ++i) { a[i] = lo[i]; a[8 + i] = hv[i]; }
  return a;
}

__device__ __forceinline__ v16bf pack16(v8bf lo, v8bf hv) {
  v16bf a;
#pragma unroll
  for (int i = 0; i < 8; ++i) { a[i] = lo[i]; a[8 + i] = hv[i]; }
  return a;
}

// ---------------- conversion kernels ----------------
__global__ void cvt_f32_bf16(const float* __restrict__ src, bf16* __restrict__ dst, int n) {
  int i = blockIdx.x * blockDim.x + threadIdx.x;
  if (i < n) dst[i] = (bf16)src[i];
}

// src [K][Nc] f32 -> dst [Nc][K] bf16  (transpose so B-fragments are contiguous)
__global__ void cvt_f32_bf16_T(const float* __restrict__ src, bf16* __restrict__ dst, int K, int Nc) {
  int i = blockIdx.x * blockDim.x + threadIdx.x;
  if (i < K * Nc) {
    int k = i / Nc;
    int n = i - k * Nc;
    dst[(size_t)n * K + k] = (bf16)src[i];
  }
}

// ---------------- QKV projection GEMM ----------------
// C[8192, 2304] = Xb[8192,768] @ WqkvT[2304,768]^T ; 32x64 tile per wave;
// scatter into Q,K (n-major) and Vt (d-major)
__global__ void gemm_qkv_kernel(const bf16* __restrict__ xb, const bf16* __restrict__ wt,
                                bf16* __restrict__ q, bf16* __restrict__ kk,
                                bf16* __restrict__ vt) {
  const int NT = (3 * DIM) / 64;  // 36 n-tiles
  int wid  = (blockIdx.x * blockDim.x + threadIdx.x) >> 5;
  int lane = threadIdx.x & 31;
  int mt = wid / NT, nt = wid - mt * NT;
  if (mt >= (BB * NN) / 32) return;
  int m0 = mt * 32, n0 = nt * 64;
  int row = lane & 15, hi = lane >> 4;

  const bf16* arow0 = xb + (size_t)(m0 + row) * DIM;
  const bf16* arow1 = xb + (size_t)(m0 + 16 + row) * DIM;
  v8f acc[2][4] = {};
  for (int k = 0; k < DIM; k += 32) {
    v16bf a0 = load_a_frag(arow0, k, hi);
    v16bf a1 = load_a_frag(arow1, k, hi);
#pragma unroll
    for (int j = 0; j < 4; ++j) {
      const bf16* brow = wt + (size_t)(n0 + j * 16 + row) * DIM;
      v16bf b = *(const v16bf*)(brow + k + hi * 16);  // 32B contiguous
      acc[0][j] = wmma_bf16(a0, b, acc[0][j]);
      acc[1][j] = wmma_bf16(a1, b, acc[1][j]);
    }
  }
#pragma unroll
  for (int j = 0; j < 4; ++j) {
    int col   = n0 + j * 16 + row;
    int which = col / DIM;
    int rem   = col - which * DIM;
    int h = rem >> 6, d = rem & 63;
#pragma unroll
    for (int u = 0; u < 2; ++u) {
#pragma unroll
      for (int r = 0; r < 8; ++r) {
        int m  = m0 + u * 16 + hi * 8 + r;
        int b_ = m >> 10, n_ = m & (NN - 1);
        bf16 val = (bf16)acc[u][j][r];
        size_t bh = (size_t)b_ * NH + h;
        if (which == 0)      q [(bh * NN + n_) * HD + d] = val;
        else if (which == 1) kk[(bh * NN + n_) * HD + d] = val;
        else                 vt[(bh * HD + d) * NN + n_] = val;
      }
    }
  }
}

// ---------------- attention (flash-style, one wave per 32-query tile) ----------------
__global__ void attn_kernel(const bf16* __restrict__ q, const bf16* __restrict__ k,
                            const bf16* __restrict__ vt, bf16* __restrict__ ao) {
  __shared__ bf16 lds_p[4][32][32];
  int wv   = threadIdx.x >> 5;
  int wid  = blockIdx.x * 4 + wv;
  int lane = threadIdx.x & 31;
  int row = lane & 15, hi = lane >> 4;
  int qt = wid & 31;   // 32 query tiles (of 32 rows) per head
  int bh = wid >> 5;   // b*NH + h
  if (bh >= BB * NH) return;
  int q0 = qt * 32;

  const bf16* Qp = q  + (size_t)bh * NN * HD;
  const bf16* Kp = k  + (size_t)bh * NN * HD;
  const bf16* Vp = vt + (size_t)bh * HD * NN;

  v16bf qa[2][2];
#pragma unroll
  for (int u = 0; u < 2; ++u) {
    const bf16* qrow = Qp + (size_t)(q0 + u * 16 + row) * HD;
    qa[u][0] = load_a_frag(qrow, 0, hi);
    qa[u][1] = load_a_frag(qrow, 32, hi);
  }

  const float c = 0.03608439182435161f * 1.4426950408889634f;  // (1/sqrt(768))*log2(e)
  float m_run[2][8], l_run[2][8], alpha[2][8];
  v8f o[2][4] = {};
#pragma unroll
  for (int u = 0; u < 2; ++u)
#pragma unroll
    for (int r = 0; r < 8; ++r) { m_run[u][r] = -1e30f; l_run[u][r] = 0.f; }

  for (int kb = 0; kb < NN; kb += 32) {
    // S = Q * K^T for a 32x32 block (2 query row-tiles x 2 key col-tiles)
    v8f s[2][2] = {};
#pragma unroll
    for (int j = 0; j < 2; ++j) {
      const bf16* krow = Kp + (size_t)(kb + j * 16 + row) * HD;
      v16bf b0 = *(const v16bf*)(krow + hi * 16);        // d = 0..31
      v16bf b1 = *(const v16bf*)(krow + 32 + hi * 16);   // d = 32..63
#pragma unroll
      for (int u = 0; u < 2; ++u) {
        s[u][j] = wmma_bf16(qa[u][0], b0, s[u][j]);
        s[u][j] = wmma_bf16(qa[u][1], b1, s[u][j]);
      }
    }
    // online softmax per query row-tile (rows span 16-lane groups)
#pragma unroll
    for (int u = 0; u < 2; ++u) {
#pragma unroll
      for (int r = 0; r < 8; ++r) {
        float mx = fmaxf(s[u][0][r], s[u][1][r]);
        mx = fmaxf(mx, __shfl_xor(mx, 8, 16));
        mx = fmaxf(mx, __shfl_xor(mx, 4, 16));
        mx = fmaxf(mx, __shfl_xor(mx, 2, 16));
        mx = fmaxf(mx, __shfl_xor(mx, 1, 16));
        float mnew = fmaxf(m_run[u][r], mx);
        float al = __builtin_exp2f((m_run[u][r] - mnew) * c);
        m_run[u][r] = mnew;
        alpha[u][r] = al;
        float e0 = __builtin_exp2f((s[u][0][r] - mnew) * c);
        float e1 = __builtin_exp2f((s[u][1][r] - mnew) * c);
        int urow = u * 16 + hi * 8 + r;
        lds_p[wv][urow][row]      = (bf16)e0;   // C-layout -> LDS
        lds_p[wv][urow][16 + row] = (bf16)e1;
        float rs = e0 + e1;
        rs += __shfl_xor(rs, 8, 16);
        rs += __shfl_xor(rs, 4, 16);
        rs += __shfl_xor(rs, 2, 16);
        rs += __shfl_xor(rs, 1, 16);
        l_run[u][r] = l_run[u][r] * al + rs;
      }
#pragma unroll
      for (int t = 0; t < 4; ++t)
#pragma unroll
        for (int r = 0; r < 8; ++r) o[u][t][r] *= alpha[u][r];
    }

    asm volatile("s_wait_dscnt 0" ::: "memory");  // wave-local LDS store->load

    // reload P in A-layout (vectorized: two 16B LDS loads per fragment)
    v16bf pa[2];
#pragma unroll
    for (int u = 0; u < 2; ++u) {
      v8bf lo = *(const v8bf*)&lds_p[wv][u * 16 + row][hi * 16];
      v8bf hv = *(const v8bf*)&lds_p[wv][u * 16 + row][hi * 16 + 8];
      pa[u] = pack16(lo, hv);
    }
    // O += P * V  (Vt is d-major, so B-fragment is contiguous; reused by both row-tiles)
#pragma unroll
    for (int t = 0; t < 4; ++t) {
      const bf16* vrow = Vp + (size_t)(t * 16 + row) * NN + kb;
      v16bf vb = *(const v16bf*)(vrow + hi * 16);
      o[0][t] = wmma_bf16(pa[0], vb, o[0][t]);
      o[1][t] = wmma_bf16(pa[1], vb, o[1][t]);
    }
  }
  // finalize: divide by l, write [b, n, h*64+d] bf16
  int h = bh % NH, b_ = bh / NH;
#pragma unroll
  for (int u = 0; u < 2; ++u)
#pragma unroll
    for (int r = 0; r < 8; ++r) l_run[u][r] = 1.f / l_run[u][r];
#pragma unroll
  for (int t = 0; t < 4; ++t) {
    int col = h * 64 + t * 16 + row;
#pragma unroll
    for (int u = 0; u < 2; ++u) {
#pragma unroll
      for (int r = 0; r < 8; ++r) {
        int n_ = q0 + u * 16 + hi * 8 + r;
        ao[((size_t)b_ * NN + n_) * DIM + col] = (bf16)(o[u][t][r] * l_run[u][r]);
      }
    }
  }
}

// ---------------- output projection GEMM (+bias, f32 out) ----------------
__global__ void gemm_out_kernel(const bf16* __restrict__ ao, const bf16* __restrict__ wt,
                                const float* __restrict__ bias, float* __restrict__ out) {
  const int NT = DIM / 64;  // 12
  int wid  = (blockIdx.x * blockDim.x + threadIdx.x) >> 5;
  int lane = threadIdx.x & 31;
  int mt = wid / NT, nt = wid - mt * NT;
  if (mt >= (BB * NN) / 32) return;
  int m0 = mt * 32, n0 = nt * 64;
  int row = lane & 15, hi = lane >> 4;

  const bf16* arow0 = ao + (size_t)(m0 + row) * DIM;
  const bf16* arow1 = ao + (size_t)(m0 + 16 + row) * DIM;
  v8f acc[2][4] = {};
  for (int k = 0; k < DIM; k += 32) {
    v16bf a0 = load_a_frag(arow0, k, hi);
    v16bf a1 = load_a_frag(arow1, k, hi);
#pragma unroll
    for (int j = 0; j < 4; ++j) {
      const bf16* brow = wt + (size_t)(n0 + j * 16 + row) * DIM;
      v16bf b = *(const v16bf*)(brow + k + hi * 16);
      acc[0][j] = wmma_bf16(a0, b, acc[0][j]);
      acc[1][j] = wmma_bf16(a1, b, acc[1][j]);
    }
  }
#pragma unroll
  for (int j = 0; j < 4; ++j) {
    int col = n0 + j * 16 + row;
    float bj = bias[col];
#pragma unroll
    for (int u = 0; u < 2; ++u) {
#pragma unroll
      for (int r = 0; r < 8; ++r) {
        int m = m0 + u * 16 + hi * 8 + r;
        out[(size_t)m * DIM + col] = acc[u][j][r] + bj;
      }
    }
  }
}

extern "C" void kernel_launch(void* const* d_in, const int* in_sizes, int n_in,
                              void* d_out, int out_size, void* d_ws, size_t ws_size,
                              hipStream_t stream) {
  const float* x     = (const float*)d_in[0];   // [8,1024,768]
  const float* w_qkv = (const float*)d_in[1];   // [768, 2304]
  const float* w_out = (const float*)d_in[2];   // [768, 768]
  const float* b_out = (const float*)d_in[3];   // [768]
  float* out = (float*)d_out;

  char* ws = (char*)d_ws;
  const size_t sz_x = (size_t)BB * NN * DIM * sizeof(bf16);        // 12.6 MB
  bf16* xb    = (bf16*)ws;               ws += sz_x;
  bf16* wqkvT = (bf16*)ws;               ws += (size_t)3 * DIM * DIM * sizeof(bf16);
  bf16* woutT = (bf16*)ws;               ws += (size_t)DIM * DIM * sizeof(bf16);
  bf16* q_ws  = (bf16*)ws;               ws += sz_x;
  bf16* k_ws  = (bf16*)ws;               ws += sz_x;
  bf16* vt_ws = (bf16*)ws;               ws += sz_x;
  bf16* ao_ws = (bf16*)ws;               ws += sz_x;

  // 1) convert inputs to bf16 (weights transposed for contiguous B-fragments)
  int nx = BB * NN * DIM;
  cvt_f32_bf16<<<(nx + 255) / 256, 256, 0, stream>>>(x, xb, nx);
  int nw1 = DIM * 3 * DIM;
  cvt_f32_bf16_T<<<(nw1 + 255) / 256, 256, 0, stream>>>(w_qkv, wqkvT, DIM, 3 * DIM);
  int nw2 = DIM * DIM;
  cvt_f32_bf16_T<<<(nw2 + 255) / 256, 256, 0, stream>>>(w_out, woutT, DIM, DIM);

  // 2) QKV projection: (8192/32) * 36 = 9216 waves, 4 waves/block
  gemm_qkv_kernel<<<9216 / 4, 128, 0, stream>>>(xb, wqkvT, q_ws, k_ws, vt_ws);

  // 3) attention: 8*12*32 = 3072 waves
  attn_kernel<<<3072 / 4, 128, 0, stream>>>(q_ws, k_ws, vt_ws, ao_ws);

  // 4) output projection: (8192/32) * 12 = 3072 waves
  gemm_out_kernel<<<3072 / 4, 128, 0, stream>>>(ao_ws, woutT, b_out, out);
}
